// BEVEncoder_81046032875794
// MI455X (gfx1250) — matrix operand order repaired and checked by reading, not compiled, not run on hardware
//
#include <hip/hip_runtime.h>
#include <math.h>

// ---------------- constants ----------------
#define BEV_HW   40000
#define EMBED    256
#define DFF      1024
#define NCAMS    6
#define HF       32
#define WF       88
#define NLAYERS  6

// ---------------- WMMA types ----------------
typedef __attribute__((ext_vector_type(16))) __bf16 v16bf;
typedef __attribute__((ext_vector_type(8)))  float  v8f;
typedef __attribute__((ext_vector_type(4)))  unsigned int v4u;

union FragAB { v4u q[2]; v16bf v; };

__device__ __forceinline__ unsigned short f2bf(float f) {
  unsigned int u = __float_as_uint(f);
  unsigned int r = u + 0x7FFFu + ((u >> 16) & 1u);   // RNE
  return (unsigned short)(r >> 16);
}

__device__ __forceinline__ v8f wmma_bf16(const FragAB& a, const FragAB& b, v8f c) {
  return __builtin_amdgcn_wmma_f32_16x16x32_bf16(false, a.v, false, b.v,
                                                 (short)0, c, false, false);
}

// ---------------- feature transpose NCHW -> NHWC ----------------
__global__ void bev_feat_nhwc_kernel(const float* __restrict__ in,
                                     float* __restrict__ out, int total) {
  int idx = blockIdx.x * blockDim.x + threadIdx.x;
  if (idx >= total) return;
  int c    = idx & 255;
  int rest = idx >> 8;
  int w    = rest % WF; rest /= WF;
  int h    = rest % HF;
  int cam  = rest / HF;
  out[idx] = in[(((size_t)cam * EMBED + c) * HF + h) * WF + w];
}

// ---------------- pack weights (KxN f32) into WMMA-B bf16 fragments ----------------
// tileIdx = (k0/32)*(N/16) + nt ; tile = 512 bf16 = [lane(32)][e(16)]
// lane l, e: K = 32*kt + 8*(l/16) + (e<8 ? e : 8+e), n = 16*nt + (l&15)
__global__ void bev_pack_b_kernel(const float* __restrict__ W,
                                  unsigned short* __restrict__ out,
                                  int K, int N) {
  int idx = blockIdx.x * blockDim.x + threadIdx.x;
  if (idx >= K * N) return;
  int tile = idx >> 9;
  int rem  = idx & 511;
  int lane = rem >> 4;
  int e    = rem & 15;
  int v = e >> 1, j = e & 1;
  int nt = tile % (N >> 4);
  int kt = tile / (N >> 4);
  int k = kt * 32 + (lane >> 4) * 8 + (e < 8 ? e : 8 + e); (void)v; (void)j;
  int n = nt * 16 + (lane & 15);
  out[idx] = f2bf(W[(size_t)k * N + n]);
}

// ---------------- projection + bilinear sampling (mean over cams) ----------------
__global__ void __launch_bounds__(256)
bev_sample_kernel(const float* __restrict__ feat /*NHWC*/,
                  const float* __restrict__ intr,
                  const float* __restrict__ extr,
                  const float* __restrict__ ref,
                  unsigned short* __restrict__ sampled) {
  __shared__ float spx[NCAMS][32];
  __shared__ float spy[NCAMS][32];
  const int p0 = blockIdx.x * 32;
  const int tid = threadIdx.x;

  if (tid < NCAMS * 32) {
    int cam = tid >> 5, p = tid & 31;
    const float* E  = extr + cam * 16;
    const float* Km = intr + cam * 9;
    float rx = ref[(size_t)(p0 + p) * 3 + 0];
    float ry = ref[(size_t)(p0 + p) * 3 + 1];
    float rz = ref[(size_t)(p0 + p) * 3 + 2];
    float dx = rx - E[3], dy = ry - E[7], dz = rz - E[11];
    float pc0 = E[0]*dx + E[4]*dy + E[8]*dz;     // pc = R^T (ref - t)
    float pc1 = E[1]*dx + E[5]*dy + E[9]*dz;
    float pc2 = E[2]*dx + E[6]*dy + E[10]*dz;
    float i0 = Km[0]*pc0 + Km[1]*pc1 + Km[2]*pc2;
    float i1 = Km[3]*pc0 + Km[4]*pc1 + Km[5]*pc2;
    float i2 = Km[6]*pc0 + Km[7]*pc1 + Km[8]*pc2;
    float z  = fmaxf(i2, 1e-5f);
    spx[cam][p] = i0 / z - 0.5f;   // algebraic simplification of ref normalization
    spy[cam][p] = i1 / z - 0.5f;
  }
  __syncthreads();

  const int c = tid;
  for (int p = 0; p < 32; ++p) {
    float acc = 0.f;
    for (int cam = 0; cam < NCAMS; ++cam) {
      float px = spx[cam][p], py = spy[cam][p];
      float x0f = floorf(px), y0f = floorf(py);
      int x0 = (int)x0f, y0 = (int)y0f;
      float wx1 = px - x0f, wy1 = py - y0f;
      float wx0 = 1.f - wx1, wy0 = 1.f - wy1;
      const float* base = feat + (size_t)cam * HF * WF * EMBED + c;
#define BEV_TAP(xi, yi, wgt)                                             \
      if ((xi) >= 0 && (xi) < WF && (yi) >= 0 && (yi) < HF)              \
        acc += (wgt) * base[((size_t)(yi) * WF + (xi)) * EMBED];
      BEV_TAP(x0,     y0,     wx0 * wy0)
      BEV_TAP(x0 + 1, y0,     wx1 * wy0)
      BEV_TAP(x0,     y0 + 1, wx0 * wy1)
      BEV_TAP(x0 + 1, y0 + 1, wx1 * wy1)
#undef BEV_TAP
    }
    sampled[(size_t)(p0 + p) * EMBED + c] = f2bf(acc * (1.0f / 6.0f));
  }
}

// ======================================================================
// Persistent 6-layer transformer kernel.
// One block owns a 16-row tile of x for the WHOLE network; x and h never
// touch HBM. A operands for FFN GEMMs come from LDS (ds_load_b128 -> wmma),
// B operands stream packed bf16 fragments from L2.
// Block = 256 threads (8 waves). Grid = 40000/16 = 2500.
// LDS: xf 16x264 f32 (16.9KB) + xh 16x256 bf16 (8KB)
//      + 32KB region multiplexed between h tile (16x1024 bf16) and the
//        f32 D-staging tile (16x264) used by the LN epilogues.  Total 57.8KB
//      -> 5 blocks (40 waves) per WGP: latency hiding via wave interleave.
// ======================================================================
__global__ void __launch_bounds__(256)
bev_layers_kernel(const unsigned short* __restrict__ sampled, // [40000,256] bf16
                  const unsigned short* __restrict__ wscap,   // packed, per-layer 256x256
                  const unsigned short* __restrict__ w1p,     // packed, per-layer 256x1024
                  const unsigned short* __restrict__ w2p,     // packed, per-layer 1024x256
                  const float* __restrict__ sca_b,  const float* __restrict__ sca_g,
                  const float* __restrict__ sca_bt,
                  const float* __restrict__ ffn_b1, const float* __restrict__ ffn_b2,
                  const float* __restrict__ ffn_g,  const float* __restrict__ ffn_bt,
                  const float* __restrict__ bev_emb, const float* __restrict__ bev_pos,
                  float* __restrict__ out) {
  __shared__ float          xf[16][264];         // x (f32), padded vs bank conflicts
  __shared__ unsigned short xh[16][256];         // x (bf16 mirror, GEMM A operand)
  __shared__ char           hraw[16 * 1024 * 2]; // h bf16 tile / aliased LN staging

  unsigned short* hs  = (unsigned short*)hraw;   // [16][1024] bf16
  float (*dtile)[264] = (float (*)[264])hraw;    // [16][264] f32 (alias)

  const int tid  = threadIdx.x;
  const int w    = tid >> 5, lane = tid & 31;
  const int half = lane >> 4, l16 = lane & 15;
  const int m0   = blockIdx.x * 16;

  const v8f vzero = {0.f,0.f,0.f,0.f,0.f,0.f,0.f,0.f};

  // ---- init: x = bev_emb + bev_pos ----
  for (int i = 0; i < 16; ++i) {
    int idx = tid + 256 * i;                  // 0..4095
    int r = idx >> 8, c = idx & 255;
    float v = bev_emb[(size_t)(m0 + r) * EMBED + c] +
              bev_pos[(size_t)(m0 + r) * EMBED + c];
    xf[r][c] = v;
    xh[r][c] = f2bf(v);
  }
  __syncthreads();

  for (int l = 0; l < NLAYERS; ++l) {
    const unsigned short* wsca = wscap + (size_t)l * EMBED * EMBED;
    const unsigned short* w1   = w1p   + (size_t)l * EMBED * DFF;
    const unsigned short* w2   = w2p   + (size_t)l * DFF * EMBED;
    const float* bsc  = sca_b  + (size_t)l * EMBED;
    const float* gsc  = sca_g  + (size_t)l * EMBED;
    const float* btsc = sca_bt + (size_t)l * EMBED;
    const float* b1   = ffn_b1 + (size_t)l * DFF;
    const float* b2   = ffn_b2 + (size_t)l * EMBED;
    const float* gfn  = ffn_g  + (size_t)l * EMBED;
    const float* btfn = ffn_bt + (size_t)l * EMBED;

    // ======== GEMM1: sca = sampled @ Wsca  (M=16, N=256, K=256) ========
    {
      v8f acc[2]; acc[0] = vzero; acc[1] = vzero;
      for (int k0 = 0; k0 < EMBED; k0 += 32) {
        FragAB a, b0, b1f;
        const unsigned short* pa =
            sampled + (size_t)(m0 + l16) * EMBED + k0 + half * 8;
        a.q[0] = *(const v4u*)pa;
        a.q[1] = *(const v4u*)(pa + 16);
        const int kt = k0 >> 5;
        const unsigned short* pb0 = wsca + (((size_t)(kt * 16 + w * 2)) << 9) + lane * 16;
        b0.q[0]  = ((const v4u*)pb0)[0];
        b0.q[1]  = ((const v4u*)pb0)[1];
        b1f.q[0] = ((const v4u*)(pb0 + 512))[0];
        b1f.q[1] = ((const v4u*)(pb0 + 512))[1];
        acc[0] = wmma_bf16(a, b0,  acc[0]);
        acc[1] = wmma_bf16(a, b1f, acc[1]);
      }
      // stage D -> dtile
      for (int j = 0; j < 2; ++j) {
        int col = (w * 2 + j) * 16 + l16;
        for (int r = 0; r < 8; ++r)
          dtile[half * 8 + r][col] = acc[j][r];
      }
      __syncthreads();
      // fused residual + LN: wave w owns rows 2w, 2w+1
      for (int rr = 0; rr < 2; ++rr) {
        int row = w * 2 + rr;
        float v[8], s = 0.f, s2 = 0.f;
        for (int i = 0; i < 8; ++i) {
          int cc = lane + 32 * i;
          float val = dtile[row][cc] + bsc[cc] + xf[row][cc];
          v[i] = val; s += val; s2 += val * val;
        }
        for (int off = 16; off >= 1; off >>= 1) {
          s  += __shfl_xor(s,  off, 32);
          s2 += __shfl_xor(s2, off, 32);
        }
        float mu  = s * (1.f / 256.f);
        float var = s2 * (1.f / 256.f) - mu * mu;
        float rs  = rsqrtf(var + 1e-5f);
        for (int i = 0; i < 8; ++i) {
          int cc = lane + 32 * i;
          float y = (v[i] - mu) * rs * gsc[cc] + btsc[cc];
          xf[row][cc] = y;
          xh[row][cc] = f2bf(y);
        }
      }
      __syncthreads();   // xh ready; dtile region free for h
    }

    // ======== GEMM2: h = gelu(x @ W1 + b1)  (M=16, N=1024, K=256) ========
    // A from LDS (xh); 8 persistent acc tiles per wave -> A reused 8x per load.
    {
      v8f acc[8];
      for (int it = 0; it < 8; ++it) acc[it] = vzero;
      for (int k0 = 0; k0 < EMBED; k0 += 32) {
        FragAB a;
        const unsigned short* pa = &xh[l16][k0 + half * 8];
        a.q[0] = *(const v4u*)pa;
        a.q[1] = *(const v4u*)(pa + 16);
        const int kt = k0 >> 5;
        for (int it = 0; it < 8; ++it) {
          int nt = it * 8 + w;
          FragAB b;
          const unsigned short* pb = w1 + (((size_t)(kt * 64 + nt)) << 9) + lane * 16;
          b.q[0] = ((const v4u*)pb)[0];
          b.q[1] = ((const v4u*)pb)[1];
          acc[it] = wmma_bf16(a, b, acc[it]);
        }
      }
      for (int it = 0; it < 8; ++it) {
        int nt = it * 8 + w;
        for (int r = 0; r < 8; ++r) {
          int row = half * 8 + r;
          int col = nt * 16 + l16;
          float vv = acc[it][r] + b1[col];
          float g  = 0.5f * vv * (1.0f + erff(vv * 0.70710678118f));
          hs[row * DFF + col] = f2bf(g);
        }
      }
      __syncthreads();   // h ready in LDS
    }

    // ======== GEMM3: x = LN(x + h @ W2 + b2)  (M=16, N=256, K=1024) ========
    {
      v8f acc[2]; acc[0] = vzero; acc[1] = vzero;
      for (int k0 = 0; k0 < DFF; k0 += 32) {
        FragAB a, b0, b1f;
        const unsigned short* pa = hs + l16 * DFF + k0 + half * 8;
        a.q[0] = *(const v4u*)pa;
        a.q[1] = *(const v4u*)(pa + 16);
        const int kt = k0 >> 5;
        const unsigned short* pb0 = w2 + (((size_t)(kt * 16 + w * 2)) << 9) + lane * 16;
        b0.q[0]  = ((const v4u*)pb0)[0];
        b0.q[1]  = ((const v4u*)pb0)[1];
        b1f.q[0] = ((const v4u*)(pb0 + 512))[0];
        b1f.q[1] = ((const v4u*)(pb0 + 512))[1];
        acc[0] = wmma_bf16(a, b0,  acc[0]);
        acc[1] = wmma_bf16(a, b1f, acc[1]);
      }
      __syncthreads();   // all h reads done -> safe to overwrite region with dtile
      for (int j = 0; j < 2; ++j) {
        int col = (w * 2 + j) * 16 + l16;
        for (int r = 0; r < 8; ++r)
          dtile[half * 8 + r][col] = acc[j][r];
      }
      __syncthreads();
      const bool last = (l == NLAYERS - 1);
      for (int rr = 0; rr < 2; ++rr) {
        int row = w * 2 + rr;
        float v[8], s = 0.f, s2 = 0.f;
        for (int i = 0; i < 8; ++i) {
          int cc = lane + 32 * i;
          float val = dtile[row][cc] + b2[cc] + xf[row][cc];
          v[i] = val; s += val; s2 += val * val;
        }
        for (int off = 16; off >= 1; off >>= 1) {
          s  += __shfl_xor(s,  off, 32);
          s2 += __shfl_xor(s2, off, 32);
        }
        float mu  = s * (1.f / 256.f);
        float var = s2 * (1.f / 256.f) - mu * mu;
        float rs  = rsqrtf(var + 1e-5f);
        for (int i = 0; i < 8; ++i) {
          int cc = lane + 32 * i;
          float y = (v[i] - mu) * rs * gfn[cc] + btfn[cc];
          if (last) {
            dtile[row][cc] = y;          // stage for transposed output write
          } else {
            xf[row][cc] = y;
            xh[row][cc] = f2bf(y);
          }
        }
      }
      __syncthreads();   // dtile stable for next layer / final write
    }
  }

  // ---- final transposed write: out[c, p] (coalesced along p) ----
  {
    int pl = tid & 15, cb = tid >> 4;          // 16 channel groups
    for (int i = 0; i < 16; ++i) {
      int cc = cb * 16 + i;
      out[(size_t)cc * BEV_HW + m0 + pl] = dtile[pl][cc];
    }
  }
}

// ---------------- host launch ----------------
extern "C" void kernel_launch(void* const* d_in, const int* in_sizes, int n_in,
                              void* d_out, int out_size, void* d_ws, size_t ws_size,
                              hipStream_t stream) {
  const float* image_feat = (const float*)d_in[0];
  const float* intr       = (const float*)d_in[1];
  const float* extr       = (const float*)d_in[2];
  const float* ref_pts    = (const float*)d_in[3];
  const float* bev_emb    = (const float*)d_in[4];
  const float* bev_pos    = (const float*)d_in[5];
  const float* sca_W      = (const float*)d_in[6];
  const float* sca_b      = (const float*)d_in[7];
  const float* sca_g      = (const float*)d_in[8];
  const float* sca_bt     = (const float*)d_in[9];
  const float* ffn_W1     = (const float*)d_in[10];
  const float* ffn_b1     = (const float*)d_in[11];
  const float* ffn_W2     = (const float*)d_in[12];
  const float* ffn_b2     = (const float*)d_in[13];
  const float* ffn_g      = (const float*)d_in[14];
  const float* ffn_bt     = (const float*)d_in[15];
  float* out = (float*)d_out;

  // workspace carve-up (all offsets 16B-aligned)
  char* ws = (char*)d_ws;
  size_t off = 0;
  float*          feat_hwc = (float*)(ws + off);          off += (size_t)NCAMS*HF*WF*EMBED*4; // 17.3 MB
  unsigned short* sampled  = (unsigned short*)(ws + off); off += (size_t)BEV_HW*EMBED*2;      // 20.5 MB
  unsigned short* wscap    = (unsigned short*)(ws + off); off += (size_t)NLAYERS*EMBED*EMBED*2;
  unsigned short* w1p      = (unsigned short*)(ws + off); off += (size_t)NLAYERS*EMBED*DFF*2;
  unsigned short* w2p      = (unsigned short*)(ws + off); off += (size_t)NLAYERS*DFF*EMBED*2;

  // 1) features NCHW -> NHWC (L2-resident afterwards)
  {
    int total = NCAMS * HF * WF * EMBED;
    bev_feat_nhwc_kernel<<<(total + 255) / 256, 256, 0, stream>>>(image_feat, feat_hwc, total);
  }
  // 2) pack all weights into WMMA-B bf16 fragment layout (L2-resident, ~7MB)
  for (int l = 0; l < NLAYERS; ++l) {
    bev_pack_b_kernel<<<(EMBED*EMBED + 255) / 256, 256, 0, stream>>>(
        sca_W + (size_t)l * EMBED * EMBED, wscap + (size_t)l * EMBED * EMBED, EMBED, EMBED);
    bev_pack_b_kernel<<<(EMBED*DFF + 255) / 256, 256, 0, stream>>>(
        ffn_W1 + (size_t)l * EMBED * DFF, w1p + (size_t)l * EMBED * DFF, EMBED, DFF);
    bev_pack_b_kernel<<<(DFF*EMBED + 255) / 256, 256, 0, stream>>>(
        ffn_W2 + (size_t)l * DFF * EMBED, w2p + (size_t)l * DFF * EMBED, DFF, EMBED);
  }
  // 3) projection + bilinear sampling, mean over cameras -> bf16 A operand
  bev_sample_kernel<<<BEV_HW / 32, 256, 0, stream>>>(feat_hwc, intr, extr, ref_pts, sampled);

  // 4) the whole 6-layer encoder in one persistent-tile kernel
  bev_layers_kernel<<<BEV_HW / 16, 256, 0, stream>>>(
      sampled, wscap, w1p, w2p,
      sca_b, sca_g, sca_bt,
      ffn_b1, ffn_b2, ffn_g, ffn_bt,
      bev_emb, bev_pos, out);

  (void)in_sizes; (void)n_in; (void)out_size; (void)ws_size;
}
